// PoincareLinear_21165598834713
// MI455X (gfx1250) — matrix operand order repaired and checked by reading, not compiled, not run on hardware
//
#include <hip/hip_runtime.h>
#include <math.h>

#define EPS_PROJ 0.004f
#define EPS_MIN  1e-15f

#define BM 128
#define BN 64
#define BK 32
#define ASTR 40   // ushort stride per LDS row (80 B pad => conflict-free b128 frag loads)
#define BSTR 40

typedef __attribute__((ext_vector_type(16))) __bf16 v16bf;
typedef __attribute__((ext_vector_type(8)))  float  v8f;

// ---- async global->LDS copy path (CDNA5), with safe fallback -------------
#if defined(__AMDGCN__) && __has_builtin(__builtin_amdgcn_global_load_async_to_lds_b128)
#define HAVE_ASYNC 1
typedef int v4i_ __attribute__((vector_size(16)));
typedef __attribute__((address_space(1))) v4i_ as1_v4i;   // global ("__device__")
typedef __attribute__((address_space(3))) v4i_ as3_v4i;   // LDS
#define CP16(dst, src) \
    __builtin_amdgcn_global_load_async_to_lds_b128((as1_v4i*)(src), (as3_v4i*)(dst), 0, 0)
#if __has_builtin(__builtin_amdgcn_s_wait_asynccnt)
#define WAIT_ASYNC() __builtin_amdgcn_s_wait_asynccnt(0)
#else
#define WAIT_ASYNC() asm volatile("s_wait_asynccnt 0x0" ::: "memory")
#endif
#else
#define HAVE_ASYNC 0
#define CP16(dst, src) (*reinterpret_cast<uint4*>(dst) = *reinterpret_cast<const uint4*>(src))
#define WAIT_ASYNC() ((void)0)
#endif

__device__ __forceinline__ unsigned short f32_bf16(float f) {
    unsigned u = __float_as_uint(f);
    u += 0x7FFFu + ((u >> 16) & 1u);           // round-to-nearest-even
    return (unsigned short)(u >> 16);
}
__device__ __forceinline__ float bf16_f32(unsigned short h) {
    return __uint_as_float(((unsigned)h) << 16);
}

// ---------------------------------------------------------------------------
// Kernel 1: per-output-column constants into colp[0..4*OUT).
// ---------------------------------------------------------------------------
__global__ __launch_bounds__(256)
void pl_prep(const float* __restrict__ wv, const float* __restrict__ wg,
             const float* __restrict__ bias, const float* __restrict__ cp,
             float* __restrict__ colp, int IN, int OUT) {
    __shared__ float red[256];
    const int j  = threadIdx.x & 63;
    const int ks = threadIdx.x >> 6;
    const int o  = blockIdx.x * 64 + j;
    float s = 0.f;
    for (int k = ks; k < IN; k += 4) {
        float v = wv[(size_t)k * OUT + o];
        s += v * v;
    }
    red[threadIdx.x] = s;
    __syncthreads();
    if (ks == 0) {
        float tot  = red[j] + red[64 + j] + red[128 + j] + red[192 + j];
        float rc   = sqrtf(cp[0]);
        float nrm  = fmaxf(sqrtf(tot), EPS_MIN);
        float drcr = 2.f * rc * bias[o];
        colp[o]           = 1.f / nrm;
        colp[OUT + o]     = coshf(drcr);
        colp[2 * OUT + o] = sinhf(drcr);
        colp[3 * OUT + o] = 2.f * wg[o] / rc;
    }
}

// ---------------------------------------------------------------------------
// Kernel 2a: convert rc*x -> bf16 hi/lo (row-major) + cx2[row] = c*||x||^2.
// ---------------------------------------------------------------------------
__global__ __launch_bounds__(256)
void pl_convert_x(const float* __restrict__ x, const float* __restrict__ cp,
                  unsigned short* __restrict__ xh, unsigned short* __restrict__ xl,
                  float* __restrict__ cx2, int IN) {
    __shared__ float red[256];
    const int r   = blockIdx.x;
    const int tid = threadIdx.x;
    const float rc = sqrtf(cp[0]);
    float s = 0.f;
    for (int base = tid * 4; base < IN; base += 256 * 4) {
        float4 v = *reinterpret_cast<const float4*>(&x[(size_t)r * IN + base]);
        float v0 = rc * v.x, v1 = rc * v.y, v2 = rc * v.z, v3 = rc * v.w;
        s += v0 * v0 + v1 * v1 + v2 * v2 + v3 * v3;
        ushort4 h, l;
        h.x = f32_bf16(v0); l.x = f32_bf16(v0 - bf16_f32(h.x));
        h.y = f32_bf16(v1); l.y = f32_bf16(v1 - bf16_f32(h.y));
        h.z = f32_bf16(v2); l.z = f32_bf16(v2 - bf16_f32(h.z));
        h.w = f32_bf16(v3); l.w = f32_bf16(v3 - bf16_f32(h.w));
        *reinterpret_cast<ushort4*>(&xh[(size_t)r * IN + base]) = h;
        *reinterpret_cast<ushort4*>(&xl[(size_t)r * IN + base]) = l;
    }
    red[tid] = s;
    __syncthreads();
    for (int off = 128; off > 0; off >>= 1) {
        if (tid < off) red[tid] += red[tid + off];
        __syncthreads();
    }
    if (tid == 0) cx2[r] = red[0];
}

// ---------------------------------------------------------------------------
// Kernel 2b: convert + transpose W -> wt[o][k] bf16 hi/lo (32x32 LDS tiles).
// ---------------------------------------------------------------------------
__global__ __launch_bounds__(256)
void pl_convert_w(const float* __restrict__ wv,
                  unsigned short* __restrict__ wth, unsigned short* __restrict__ wtl,
                  int IN, int OUT) {
    __shared__ unsigned short th[32][33];
    __shared__ unsigned short tl[32][33];
    const int o0 = blockIdx.x * 32;
    const int k0 = blockIdx.y * 32;
    const int c  = threadIdx.x & 31;
    const int r4 = threadIdx.x >> 5;     // 0..7
#pragma unroll
    for (int i = 0; i < 4; ++i) {
        const int r = r4 * 4 + i;
        float w = wv[(size_t)(k0 + r) * OUT + o0 + c];
        unsigned short h = f32_bf16(w);
        th[r][c] = h;
        tl[r][c] = f32_bf16(w - bf16_f32(h));
    }
    __syncthreads();
#pragma unroll
    for (int i = 0; i < 4; ++i) {
        const int r = r4 * 4 + i;
        wth[(size_t)(o0 + r) * IN + k0 + c] = th[c][r];
        wtl[(size_t)(o0 + r) * IN + k0 + c] = tl[c][r];
    }
}

// ---------------------------------------------------------------------------
// Shared pieces for the GEMMs.
// ---------------------------------------------------------------------------
__device__ __forceinline__ void pl_stage(const unsigned short* __restrict__ xh,
                                         const unsigned short* __restrict__ xl,
                                         const unsigned short* __restrict__ wth,
                                         const unsigned short* __restrict__ wtl,
                                         unsigned short* AhB, unsigned short* AlB,
                                         unsigned short* BhB, unsigned short* BlB,
                                         int tid, int rbase, int nbase, int IN, int k0) {
    // A tile: 128 rows x 64B (BK bf16), 512 x 16B chunks per precision
#pragma unroll
    for (int s = 0; s < 2; ++s) {
        const int cid = tid + 256 * s;
        const int row = cid >> 2;
        const int q   = (cid & 3) * 16;     // byte offset in row chunk
        const char* sh = (const char*)(&xh[(size_t)(rbase + row) * IN + k0]) + q;
        const char* sl = (const char*)(&xl[(size_t)(rbase + row) * IN + k0]) + q;
        CP16((char*)AhB + row * 80 + q, sh);
        CP16((char*)AlB + row * 80 + q, sl);
    }
    // B tile: 64 cols x 64B, 256 x 16B chunks per precision
    {
        const int n = tid >> 2;
        const int q = (tid & 3) * 16;
        const char* sh = (const char*)(&wth[(size_t)(nbase + n) * IN + k0]) + q;
        const char* sl = (const char*)(&wtl[(size_t)(nbase + n) * IN + k0]) + q;
        CP16((char*)BhB + n * 80 + q, sh);
        CP16((char*)BlB + n * 80 + q, sl);
    }
}

__device__ __forceinline__ void pl_mma(const unsigned short* AhB, const unsigned short* AlB,
                                       const unsigned short* BhB, const unsigned short* BlB,
                                       int lane, int wm, int wn, v8f (&acc)[2][2]) {
    const int kbA = (lane >> 4) * 8;    // lanes 0-15: K 0..7/16..23; 16-31: 8..15/24..31
    const int kbB = (lane >> 4) * 8;    // lanes 0-15: K 0..7/16..23; 16-31: 8..15/24..31
    union F { v16bf v; uint4 q[2]; };
    F a_h[2], a_l[2], b_h[2], b_l[2];
#pragma unroll
    for (int fm = 0; fm < 2; ++fm) {
        const int row = wm * 32 + fm * 16 + (lane & 15);
        a_h[fm].q[0] = *reinterpret_cast<const uint4*>(&AhB[row * ASTR + kbA]);
        a_h[fm].q[1] = *reinterpret_cast<const uint4*>(&AhB[row * ASTR + kbA + 16]);
        a_l[fm].q[0] = *reinterpret_cast<const uint4*>(&AlB[row * ASTR + kbA]);
        a_l[fm].q[1] = *reinterpret_cast<const uint4*>(&AlB[row * ASTR + kbA + 16]);
    }
#pragma unroll
    for (int fn = 0; fn < 2; ++fn) {
        const int col = wn * 32 + fn * 16 + (lane & 15);
        b_h[fn].q[0] = *reinterpret_cast<const uint4*>(&BhB[col * BSTR + kbB]);
        b_h[fn].q[1] = *reinterpret_cast<const uint4*>(&BhB[col * BSTR + kbB + 16]);
        b_l[fn].q[0] = *reinterpret_cast<const uint4*>(&BlB[col * BSTR + kbB]);
        b_l[fn].q[1] = *reinterpret_cast<const uint4*>(&BlB[col * BSTR + kbB + 16]);
    }
#pragma unroll
    for (int fm = 0; fm < 2; ++fm)
#pragma unroll
        for (int fn = 0; fn < 2; ++fn) {
            acc[fm][fn] = __builtin_amdgcn_wmma_f32_16x16x32_bf16(
                false, a_h[fm].v, false, b_h[fn].v, (short)0, acc[fm][fn], false, false);
            acc[fm][fn] = __builtin_amdgcn_wmma_f32_16x16x32_bf16(
                false, a_h[fm].v, false, b_l[fn].v, (short)0, acc[fm][fn], false, false);
            acc[fm][fn] = __builtin_amdgcn_wmma_f32_16x16x32_bf16(
                false, a_l[fm].v, false, b_h[fn].v, (short)0, acc[fm][fn], false, false);
        }
}

__device__ __forceinline__ void pl_epilogue(const v8f (&acc)[2][2], const float* colp,
                                            const float* cx2row, float rc,
                                            float* __restrict__ out,
                                            int rbase, int nbase, int lane, int wm, int wn,
                                            int OUT) {
    const float* inv_norm = colp;
    const float* coshd    = colp + OUT;
    const float* sinhd    = colp + 2 * OUT;
    const float* ggp      = colp + 3 * OUT;
#pragma unroll
    for (int fm = 0; fm < 2; ++fm) {
        const int rloc = wm * 32 + fm * 16 + 8 * (lane >> 4);
#pragma unroll
        for (int fn = 0; fn < 2; ++fn) {
            const int o = nbase + wn * 32 + fn * 16 + (lane & 15);
            const float inv = inv_norm[o], ch = coshd[o], sh = sinhd[o], gg = ggp[o];
#pragma unroll
            for (int j = 0; j < 8; ++j) {
                const int r   = rloc + j;
                const float cx2 = cx2row[r];
                const float g   = acc[fm][fn][j] * inv;
                const float num = 2.f * g * ch - (1.f + cx2) * sh;
                const float den = fmaxf(1.f - cx2, EPS_MIN);
                const float mlr = gg * asinhf(num / den);
                const float t   = sinhf(rc * mlr) / rc;
                out[(size_t)(rbase + r) * OUT + o] = t;
            }
        }
    }
}

// ---------------------------------------------------------------------------
// Kernel 3 (fast path): double-buffered bf16x3 GEMM on preconverted operands.
// Staging is pure 16B copies (async global->LDS when available).
// ---------------------------------------------------------------------------
__global__ __launch_bounds__(256)
void pl_gemm_pre(const unsigned short* __restrict__ xh, const unsigned short* __restrict__ xl,
                 const unsigned short* __restrict__ wth, const unsigned short* __restrict__ wtl,
                 const float* __restrict__ colp, const float* __restrict__ cx2g,
                 const float* __restrict__ cp, float* __restrict__ out,
                 int N, int IN, int OUT) {
    __shared__ __align__(16) unsigned short Ah[2][BM * ASTR];
    __shared__ __align__(16) unsigned short Al[2][BM * ASTR];
    __shared__ __align__(16) unsigned short Bh[2][BN * BSTR];
    __shared__ __align__(16) unsigned short Bl[2][BN * BSTR];

    const int tid  = threadIdx.x;
    const int lane = tid & 31;
    const int wave = tid >> 5;
    const int wm   = wave & 3;
    const int wn   = wave >> 2;
    const int rbase = blockIdx.y * BM;
    const int nbase = blockIdx.x * BN;
    const float rc = sqrtf(cp[0]);

    v8f acc[2][2];
#pragma unroll
    for (int a = 0; a < 2; ++a)
#pragma unroll
        for (int b = 0; b < 2; ++b)
#pragma unroll
            for (int j = 0; j < 8; ++j) acc[a][b][j] = 0.f;

    const int KT = IN / BK;
    pl_stage(xh, xl, wth, wtl, Ah[0], Al[0], Bh[0], Bl[0], tid, rbase, nbase, IN, 0);
    WAIT_ASYNC();
    __syncthreads();

    for (int kt = 0; kt < KT; ++kt) {
        const int cur = kt & 1;
        if (kt + 1 < KT)
            pl_stage(xh, xl, wth, wtl, Ah[1 - cur], Al[1 - cur], Bh[1 - cur], Bl[1 - cur],
                     tid, rbase, nbase, IN, (kt + 1) * BK);
        pl_mma(Ah[cur], Al[cur], Bh[cur], Bl[cur], lane, wm, wn, acc);
        WAIT_ASYNC();
        __syncthreads();
    }

    pl_epilogue(acc, colp, &cx2g[rbase], rc, out, rbase, nbase, lane, wm, wn, OUT);
}

// ---------------------------------------------------------------------------
// Kernel 3 (fallback path, small workspace): fused in-loop conversion GEMM.
// ---------------------------------------------------------------------------
__global__ __launch_bounds__(256)
void pl_gemm_fused(const float* __restrict__ x, const float* __restrict__ wv,
                   const float* __restrict__ colp, const float* __restrict__ cp,
                   float* __restrict__ out, int N, int IN, int OUT) {
    __shared__ __align__(16) unsigned short Ah[BM * ASTR];
    __shared__ __align__(16) unsigned short Al[BM * ASTR];
    __shared__ __align__(16) unsigned short Bh[BN * BSTR];
    __shared__ __align__(16) unsigned short Bl[BN * BSTR];
    __shared__ float cxs[BM];

    const int tid  = threadIdx.x;
    const int lane = tid & 31;
    const int wave = tid >> 5;
    const int wm   = wave & 3;
    const int wn   = wave >> 2;
    const int rbase = blockIdx.y * BM;
    const int nbase = blockIdx.x * BN;
    const float rc = sqrtf(cp[0]);

    if (tid < BM) cxs[tid] = 0.f;

    const int ar  = tid >> 3;
    const int ak  = (tid & 7) * 4;
    const int bn  = tid & 63;
    const int bk0 = tid >> 6;

    float accsq[4] = {0.f, 0.f, 0.f, 0.f};
    v8f acc[2][2];
#pragma unroll
    for (int a = 0; a < 2; ++a)
#pragma unroll
        for (int b = 0; b < 2; ++b)
#pragma unroll
            for (int j = 0; j < 8; ++j) acc[a][b][j] = 0.f;

    const int KT = IN / BK;
    for (int kt = 0; kt < KT; ++kt) {
        const int k0 = kt * BK;
        __syncthreads();
#pragma unroll
        for (int i = 0; i < 4; ++i) {
            const int row = ar + 32 * i;
            const float4 xv =
                *reinterpret_cast<const float4*>(&x[(size_t)(rbase + row) * IN + k0 + ak]);
            float v0 = rc * xv.x, v1 = rc * xv.y, v2 = rc * xv.z, v3 = rc * xv.w;
            accsq[i] += v0 * v0 + v1 * v1 + v2 * v2 + v3 * v3;
            unsigned short h0 = f32_bf16(v0), h1 = f32_bf16(v1);
            unsigned short h2 = f32_bf16(v2), h3 = f32_bf16(v3);
            unsigned short* ah = &Ah[row * ASTR + ak];
            unsigned short* al = &Al[row * ASTR + ak];
            ah[0] = h0; ah[1] = h1; ah[2] = h2; ah[3] = h3;
            al[0] = f32_bf16(v0 - bf16_f32(h0));
            al[1] = f32_bf16(v1 - bf16_f32(h1));
            al[2] = f32_bf16(v2 - bf16_f32(h2));
            al[3] = f32_bf16(v3 - bf16_f32(h3));
        }
#pragma unroll
        for (int i = 0; i < 8; ++i) {
            const int kk = bk0 + 4 * i;
            float w = wv[(size_t)(k0 + kk) * OUT + nbase + bn];
            unsigned short h = f32_bf16(w);
            Bh[bn * BSTR + kk] = h;
            Bl[bn * BSTR + kk] = f32_bf16(w - bf16_f32(h));
        }
        __syncthreads();
        pl_mma(Ah, Al, Bh, Bl, lane, wm, wn, acc);
    }

    __syncthreads();
#pragma unroll
    for (int i = 0; i < 4; ++i) atomicAdd(&cxs[ar + 32 * i], accsq[i]);
    __syncthreads();

    pl_epilogue(acc, colp, cxs, rc, out, rbase, nbase, lane, wm, wn, OUT);
}

// ---------------------------------------------------------------------------
// Kernel 4: per-row finalize (gyro-half map + projection), in place.
// ---------------------------------------------------------------------------
__global__ __launch_bounds__(256)
void pl_final(float* __restrict__ out, const float* __restrict__ cp, int OUT) {
    __shared__ float red[256];
    const int r   = blockIdx.x;
    const int tid = threadIdx.x;
    float s = 0.f;
    for (int base = tid * 4; base < OUT; base += 256 * 4) {
        float4 v = *reinterpret_cast<const float4*>(&out[(size_t)r * OUT + base]);
        s += v.x * v.x + v.y * v.y + v.z * v.z + v.w * v.w;
    }
    red[tid] = s;
    __syncthreads();
    for (int off = 128; off > 0; off >>= 1) {
        if (tid < off) red[tid] += red[tid + off];
        __syncthreads();
    }
    const float S  = red[0];
    const float c  = cp[0];
    const float sc = 1.f / (1.f + sqrtf(1.f + c * S));
    const float ynorm = fmaxf(sc * sqrtf(S), EPS_MIN);
    const float k = -c;
    const float maxnorm =
        (k < 0.f) ? (1.f - EPS_PROJ) / sqrtf(fmaxf(fabsf(k), EPS_MIN)) : 1e15f;
    const float scale = (ynorm > maxnorm) ? sc * (maxnorm / ynorm) : sc;
    for (int base = tid * 4; base < OUT; base += 256 * 4) {
        float4 v = *reinterpret_cast<const float4*>(&out[(size_t)r * OUT + base]);
        v.x *= scale; v.y *= scale; v.z *= scale; v.w *= scale;
        *reinterpret_cast<float4*>(&out[(size_t)r * OUT + base]) = v;
    }
}

// ---------------------------------------------------------------------------
extern "C" void kernel_launch(void* const* d_in, const int* in_sizes, int n_in,
                              void* d_out, int out_size, void* d_ws, size_t ws_size,
                              hipStream_t stream) {
    const float* x    = (const float*)d_in[0];
    const float* wv   = (const float*)d_in[1];
    const float* wg   = (const float*)d_in[2];
    const float* bias = (const float*)d_in[3];
    const float* cp   = (const float*)d_in[4];

    const int OUT = in_sizes[2];              // 1024
    const int IN  = in_sizes[1] / OUT;        // 1024
    const int N   = in_sizes[0] / IN;         // 32768

    float* out = (float*)d_out;
    char*  ws  = (char*)d_ws;

    size_t off = 0;
    auto take = [&](size_t bytes) {
        off = (off + 255) & ~(size_t)255;
        size_t o = off;
        off += bytes;
        return o;
    };
    const size_t o_colp = take((size_t)4 * OUT * sizeof(float));
    const size_t o_cx2  = take((size_t)N * sizeof(float));
    const size_t o_wth  = take((size_t)OUT * IN * sizeof(unsigned short));
    const size_t o_wtl  = take((size_t)OUT * IN * sizeof(unsigned short));
    const size_t o_xh   = take((size_t)N * IN * sizeof(unsigned short));
    const size_t o_xl   = take((size_t)N * IN * sizeof(unsigned short));

    float* colp = (float*)(ws + o_colp);

    pl_prep<<<dim3(OUT / 64), 256, 0, stream>>>(wv, wg, bias, cp, colp, IN, OUT);

    if (ws_size >= off) {
        unsigned short* wth = (unsigned short*)(ws + o_wth);
        unsigned short* wtl = (unsigned short*)(ws + o_wtl);
        unsigned short* xh  = (unsigned short*)(ws + o_xh);
        unsigned short* xl  = (unsigned short*)(ws + o_xl);
        float* cx2 = (float*)(ws + o_cx2);
        pl_convert_w<<<dim3(OUT / 32, IN / 32), 256, 0, stream>>>(wv, wth, wtl, IN, OUT);
        pl_convert_x<<<dim3(N), 256, 0, stream>>>(x, cp, xh, xl, cx2, IN);
        pl_gemm_pre<<<dim3(OUT / BN, N / BM), 256, 0, stream>>>(
            xh, xl, wth, wtl, colp, cx2, cp, out, N, IN, OUT);
    } else {
        pl_gemm_fused<<<dim3(OUT / BN, N / BM), 256, 0, stream>>>(
            x, wv, colp, cp, out, N, IN, OUT);
    }

    pl_final<<<dim3(N), 256, 0, stream>>>(out, cp, OUT);
}
